// ElasticMoEModel_6571299963110
// MI455X (gfx1250) — matrix operand order repaired and checked by reading, not compiled, not run on hardware
//
#include <hip/hip_runtime.h>
#include <math.h>

// ---------------------------------------------------------------------------
// ElasticMoE forward for MI455X (gfx1250).
// - All large GEMMs on v_wmma_f32_16x16x32_bf16 (4-wave blocks, 64x64 tile,
//   B staged via LDS transposed bf16, conflict-free ds_load_b128 fragments).
// - Weights pre-converted to bf16 once (half the weight traffic, no cvt in
//   the k-loop); activations produced in bf16 by the upstream epilogues.
// - Gathered top-2 MoE (4x FLOP reduction vs dense-all-experts).
// ---------------------------------------------------------------------------

typedef __attribute__((ext_vector_type(16))) __bf16 v16bf;
typedef __attribute__((ext_vector_type(8)))  __bf16 v8bf;
typedef __attribute__((ext_vector_type(8)))  float  v8f;

#define BT   2048   // tokens (batch)
#define DD   1024   // model dim
#define HH   2048   // expert hidden dim
#define EE   8      // experts
#define NBLKC 6
#define NCLSC 10
#define EPSF 1e-5f

// LDS B tile: [64 n][32 k] bf16, row stride 40 (80B: 16B-aligned, bank-conflict free)
#define LDSB_STRIDE 40

__device__ __forceinline__ float gelu_exact(float x) {
    return 0.5f * x * (1.0f + erff(x * 0.70710678118654752f));
}

// ---------------------------------------------------------------- f32 -> bf16
__global__ void cvt_bf16_k(const float* __restrict__ in, __bf16* __restrict__ out, long n4)
{
    long i = (long)blockIdx.x * blockDim.x + threadIdx.x;
    if (i >= n4) return;
    float4 q = ((const float4*)in)[i];
    out[i * 4 + 0] = (__bf16)q.x;
    out[i * 4 + 1] = (__bf16)q.y;
    out[i * 4 + 2] = (__bf16)q.z;
    out[i * 4 + 3] = (__bf16)q.w;
}

// ---------------------------------------------------------------- conv+bn+gelu
__global__ void conv_bn_gelu(const float* __restrict__ in, const float* __restrict__ w,
                             const float* __restrict__ cb,
                             const float* __restrict__ g, const float* __restrict__ b,
                             const float* __restrict__ m, const float* __restrict__ v,
                             float* __restrict__ out, int N, int Cin, int Cout)
{
    const int HW = 32;
    int idx = blockIdx.x * blockDim.x + threadIdx.x;
    int total = N * Cout * HW * HW;
    if (idx >= total) return;
    int x  = idx & (HW - 1);
    int y  = (idx / HW) & (HW - 1);
    int co = (idx / (HW * HW)) % Cout;
    int n  = idx / (HW * HW * Cout);
    float acc = cb[co];
    const float* wp = w + co * Cin * 9;
    for (int ci = 0; ci < Cin; ++ci) {
        const float* ip = in + ((n * Cin + ci) * HW) * HW;
        #pragma unroll
        for (int ky = 0; ky < 3; ++ky) {
            int yy = y + ky - 1;
            if (yy < 0 || yy >= HW) continue;
            #pragma unroll
            for (int kx = 0; kx < 3; ++kx) {
                int xx = x + kx - 1;
                if (xx < 0 || xx >= HW) continue;
                acc += ip[yy * HW + xx] * wp[ci * 9 + ky * 3 + kx];
            }
        }
    }
    float s = g[co] * rsqrtf(v[co] + EPSF);
    float o = (acc - m[co]) * s + b[co];
    out[idx] = gelu_exact(o);
}

// ------------------------------------------------- avg pool 8x8 -> bf16 flatten
__global__ void avgpool_flatten(const float* __restrict__ in, __bf16* __restrict__ out, int N)
{
    int idx = blockIdx.x * blockDim.x + threadIdx.x;
    int total = N * 128 * 4 * 4;
    if (idx >= total) return;
    int j = idx & 3;
    int i = (idx >> 2) & 3;
    int c = (idx >> 4) & 127;
    int n = idx >> 11;
    const float* ip = in + ((n * 128 + c) * 32) * 32;
    float s = 0.f;
    #pragma unroll
    for (int u = 0; u < 8; ++u)
        #pragma unroll
        for (int w = 0; w < 8; ++w)
            s += ip[(i * 8 + u) * 32 + (j * 8 + w)];
    out[n * 2048 + c * 16 + i * 4 + j] = (__bf16)(s * (1.0f / 64.0f));
}

// ---------------------------------------------------- layernorm (f32 in, bf16 out)
__global__ void layernorm_k(const float* __restrict__ in, const float* __restrict__ g,
                            const float* __restrict__ b, __bf16* __restrict__ out, int D)
{
    __shared__ float red[256];
    int row = blockIdx.x;
    const float* x = in + (size_t)row * D;
    int tid = threadIdx.x;

    float s = 0.f;
    for (int d = tid; d < D; d += 256) s += x[d];
    red[tid] = s; __syncthreads();
    for (int off = 128; off > 0; off >>= 1) {
        if (tid < off) red[tid] += red[tid + off];
        __syncthreads();
    }
    float mean = red[0] / D;
    __syncthreads();

    float vs = 0.f;
    for (int d = tid; d < D; d += 256) { float t = x[d] - mean; vs += t * t; }
    red[tid] = vs; __syncthreads();
    for (int off = 128; off > 0; off >>= 1) {
        if (tid < off) red[tid] += red[tid + off];
        __syncthreads();
    }
    float rstd = rsqrtf(red[0] / D + EPSF);

    for (int d = tid; d < D; d += 256)
        out[(size_t)row * D + d] = (__bf16)((x[d] - mean) * rstd * g[d] + b[d]);
}

// ---------------------------------------------------------------- router
__global__ void router_k(const __bf16* __restrict__ r, const float* __restrict__ rw,
                         const float* __restrict__ rb, float* __restrict__ probs,
                         int* __restrict__ topi, float* __restrict__ topg)
{
    int t = blockIdx.x * blockDim.x + threadIdx.x;
    if (t >= BT) return;
    float lg[EE];
    #pragma unroll
    for (int e = 0; e < EE; ++e) lg[e] = rb[e];
    const __bf16* rr = r + (size_t)t * DD;
    for (int d = 0; d < DD; ++d) {
        float rv = (float)rr[d];
        const float* wp = rw + d * EE;
        #pragma unroll
        for (int e = 0; e < EE; ++e) lg[e] += rv * wp[e];
    }
    float mx = lg[0];
    #pragma unroll
    for (int e = 1; e < EE; ++e) mx = fmaxf(mx, lg[e]);
    float den = 0.f, p[EE];
    #pragma unroll
    for (int e = 0; e < EE; ++e) { p[e] = __expf(lg[e] - mx); den += p[e]; }
    #pragma unroll
    for (int e = 0; e < EE; ++e) { p[e] /= den; probs[t * EE + e] = p[e]; }
    int i1 = 0; float v1 = p[0];
    #pragma unroll
    for (int e = 1; e < EE; ++e) if (p[e] > v1) { v1 = p[e]; i1 = e; }
    int i2 = (i1 == 0) ? 1 : 0; float v2 = p[i2];
    #pragma unroll
    for (int e = 0; e < EE; ++e) if (e != i1 && p[e] > v2) { v2 = p[e]; i2 = e; }
    float gs = v1 + v2;
    topi[t * 2 + 0] = i1; topi[t * 2 + 1] = i2;
    topg[t * 2 + 0] = v1 / gs; topg[t * 2 + 1] = v2 / gs;
}

// ---------------------------------------------------------------- gather setup
__global__ void zero_counts_k(int* counts) { if (threadIdx.x < EE) counts[threadIdx.x] = 0; }

__global__ void count_k(const int* __restrict__ topi, int* __restrict__ counts)
{
    int t = blockIdx.x * blockDim.x + threadIdx.x;
    if (t >= BT) return;
    atomicAdd(&counts[topi[t * 2 + 0]], 1);
    atomicAdd(&counts[topi[t * 2 + 1]], 1);
}

__global__ void scan_k(const int* __restrict__ counts, int* __restrict__ offs, int* __restrict__ cur)
{
    if (threadIdx.x == 0) {
        int s = 0;
        for (int e = 0; e < EE; ++e) { offs[e] = s; cur[e] = s; s += counts[e]; }
    }
}

__global__ void scatter_k(const int* __restrict__ topi, const float* __restrict__ topg,
                          int* __restrict__ cur, int* __restrict__ list, float* __restrict__ glist)
{
    int t = blockIdx.x * blockDim.x + threadIdx.x;
    if (t >= BT) return;
    #pragma unroll
    for (int k = 0; k < 2; ++k) {
        int e = topi[t * 2 + k];
        int pos = atomicAdd(&cur[e], 1);
        list[pos]  = t;
        glist[pos] = topg[t * 2 + k];
    }
}

// ---------------------------------------------------------------- aux loss
__global__ void aux_reduce_k(const float* __restrict__ probs, const int* __restrict__ counts,
                             float* __restrict__ prod)
{
    __shared__ float red[256];
    int e = blockIdx.x, tid = threadIdx.x;
    float s = 0.f;
    for (int t = tid; t < BT; t += 256) s += probs[t * EE + e];
    red[tid] = s; __syncthreads();
    for (int off = 128; off > 0; off >>= 1) {
        if (tid < off) red[tid] += red[tid + off];
        __syncthreads();
    }
    if (tid == 0) prod[e] = ((float)counts[e] / (float)BT) * (red[0] / (float)BT);
}

__global__ void aux_accum_k(const float* __restrict__ prod, float* __restrict__ aux)
{
    if (threadIdx.x == 0) {
        float s = 0.f;
        for (int e = 0; e < EE; ++e) s += prod[e];
        *aux += (float)EE * s;
    }
}

__global__ void init_aux_k(float* aux) { if (threadIdx.x == 0) *aux = 0.f; }
__global__ void write_aux_k(const float* aux, float* out) { if (threadIdx.x == 0) *out = *aux; }

// ---------------------------------------------------------------- WMMA helpers
// A fragment 16x32 bf16: lanes 0-15 -> kb=0; lanes 16-31 -> kb=8.
// elements 0..7 = K kb+0..7 ; 8..15 = K kb+16..23. Two 16B v8bf loads.
__device__ __forceinline__ v16bf load_a_frag_bf(const __bf16* __restrict__ Arow, int lane)
{
    int kb = (lane < 16) ? 0 : 8;
    const v8bf* p = (const v8bf*)(Arow + kb);   // 16B aligned
    v8bf lo = p[0];   // kb+0..7
    v8bf hi = p[2];   // kb+16..23
    v16bf a;
    #pragma unroll
    for (int j = 0; j < 8; ++j) { a[j] = lo[j]; a[8 + j] = hi[j]; }
    return a;
}

// B fragment from transposed LDS tile [n][k]: lane = column n (mod 16);
// lanes 0-15 hold K=0..7,16..23; lanes 16-31 hold K=8..15,24..31 (matches A).
__device__ __forceinline__ v16bf read_b_frag_lds(const __bf16* __restrict__ ldsb,
                                                 int s, int lane)
{
    int n  = s * 16 + (lane & 15);
    int kb = (lane < 16) ? 0 : 8;
    const v8bf* p = (const v8bf*)(ldsb + n * LDSB_STRIDE + kb);   // 16B aligned
    v8bf lo = p[0];
    v8bf hi = p[2];
    v16bf b;
    #pragma unroll
    for (int j = 0; j < 8; ++j) { b[j] = lo[j]; b[8 + j] = hi[j]; }
    return b;
}

#define WMMA_BF16(A, B, C) \
    __builtin_amdgcn_wmma_f32_16x16x32_bf16(false, (A), false, (B), (short)0, (C), false, false)

// Shared 4-wave GEMM core (all-bf16 operands): 64x64 macro tile per block.
// Arow = per-lane A row pointer; Bt0 = B + n0 column offset applied (bf16, ldb cols).
__device__ __forceinline__ void gemm_core_bf(const __bf16* __restrict__ Arow,
                                             const __bf16* __restrict__ Bt0,
                                             int ldb, int K,
                                             __bf16* __restrict__ ldsb, v8f acc[4])
{
    int tid  = threadIdx.x;
    int lane = tid & 31;
    for (int k0 = 0; k0 < K; k0 += 32) {
        __syncthreads();   // previous-iteration LDS reads complete
        // cooperative staging of B[32k x 64n] bf16 -> LDS [n][k] (pure moves, no cvt)
        #pragma unroll
        for (int it = 0; it < 2; ++it) {
            int l  = tid + it * 128;          // 0..255
            int k  = l >> 3;                  // 0..31
            int n8 = (l & 7) << 3;            // 0,8,..,56
            v8bf q = *(const v8bf*)(Bt0 + (size_t)(k0 + k) * ldb + n8);
            #pragma unroll
            for (int j = 0; j < 8; ++j)
                ldsb[(n8 + j) * LDSB_STRIDE + k] = q[j];
        }
        __syncthreads();
        __builtin_prefetch(Arow + k0 + 128, 0, 1);
        v16bf a = load_a_frag_bf(Arow + k0, lane);
        acc[0] = WMMA_BF16(a, read_b_frag_lds(ldsb, 0, lane), acc[0]);
        acc[1] = WMMA_BF16(a, read_b_frag_lds(ldsb, 1, lane), acc[1]);
        acc[2] = WMMA_BF16(a, read_b_frag_lds(ldsb, 2, lane), acc[2]);
        acc[3] = WMMA_BF16(a, read_b_frag_lds(ldsb, 3, lane), acc[3]);
    }
}

// ---------------------------------------------------------------- fc GEMM (gelu)
// C[M,N] = gelu(A[M,K] * B[K,N] + bias[N]); A,B bf16; C f32.
__global__ __launch_bounds__(128) void gemm_bias_gelu(const __bf16* __restrict__ A,
                                                      const __bf16* __restrict__ Bw,
                                                      const float* __restrict__ bias,
                                                      float* __restrict__ C,
                                                      int M, int N, int K)
{
    __shared__ __bf16 ldsb[64 * LDSB_STRIDE];
    int lane = threadIdx.x & 31;
    int wid  = threadIdx.x >> 5;
    int n0   = blockIdx.x * 64;
    int m0   = blockIdx.y * 64 + wid * 16;
    int mrow = m0 + (lane & 15);
    const __bf16* Arow = A + (size_t)mrow * K;
    v8f acc[4] = {v8f{}, v8f{}, v8f{}, v8f{}};
    gemm_core_bf(Arow, Bw + n0, N, K, ldsb, acc);

    int n = lane & 15;
    int mbase = (lane < 16) ? 0 : 8;
    #pragma unroll
    for (int s = 0; s < 4; ++s) {
        int nc = n0 + s * 16 + n;
        float bb = bias[nc];
        #pragma unroll
        for (int i = 0; i < 8; ++i) {
            int m = m0 + mbase + i;
            C[(size_t)m * N + nc] = gelu_exact(acc[s][i] + bb);
        }
    }
}

// --------------------------------------------- gathered expert GEMM1 (r @ w1, gelu)
__global__ __launch_bounds__(128) void moe_gemm1(const __bf16* __restrict__ r,
                                                 const __bf16* __restrict__ w1,
                                                 const float* __restrict__ b1,
                                                 const int* __restrict__ list,
                                                 const int* __restrict__ offs,
                                                 const int* __restrict__ counts,
                                                 __bf16* __restrict__ ehg)
{
    __shared__ __bf16 ldsb[64 * LDSB_STRIDE];
    int e   = blockIdx.z;
    int cnt = counts[e];
    int mb0 = blockIdx.y * 64;
    if (mb0 >= cnt) return;                 // uniform across whole block
    int lane = threadIdx.x & 31;
    int wid  = threadIdx.x >> 5;
    int n0   = blockIdx.x * 64;
    int base = offs[e];
    int m0   = mb0 + wid * 16;
    int grow = base + min(m0 + (lane & 15), cnt - 1);   // clamped gather (dummy ok)
    int tok  = list[grow];
    const __bf16* Arow = r + (size_t)tok * DD;
    const __bf16* Bw   = w1 + (size_t)e * DD * HH;
    v8f acc[4] = {v8f{}, v8f{}, v8f{}, v8f{}};
    gemm_core_bf(Arow, Bw + n0, HH, DD, ldsb, acc);

    int n = lane & 15;
    int mbase = (lane < 16) ? 0 : 8;
    #pragma unroll
    for (int s = 0; s < 4; ++s) {
        int nc = n0 + s * 16 + n;
        float bb = b1[e * HH + nc];
        #pragma unroll
        for (int i = 0; i < 8; ++i) {
            int m = m0 + mbase + i;
            if (m < cnt)
                ehg[(size_t)(base + m) * HH + nc] = (__bf16)gelu_exact(acc[s][i] + bb);
        }
    }
}

// ----------------------------------- gathered expert GEMM2 (eh @ w2, gate, scatter)
__global__ __launch_bounds__(128) void moe_gemm2(const __bf16* __restrict__ ehg,
                                                 const __bf16* __restrict__ w2,
                                                 const float* __restrict__ b2,
                                                 const int* __restrict__ list,
                                                 const float* __restrict__ glist,
                                                 const int* __restrict__ offs,
                                                 const int* __restrict__ counts,
                                                 float* __restrict__ hOut)
{
    __shared__ __bf16 ldsb[64 * LDSB_STRIDE];
    int e   = blockIdx.z;
    int cnt = counts[e];
    int mb0 = blockIdx.y * 64;
    if (mb0 >= cnt) return;                 // uniform across whole block
    int lane = threadIdx.x & 31;
    int wid  = threadIdx.x >> 5;
    int n0   = blockIdx.x * 64;
    int base = offs[e];
    int m0   = mb0 + wid * 16;
    int arow = base + min(m0 + (lane & 15), cnt - 1);
    const __bf16* Arow = ehg + (size_t)arow * HH;
    const __bf16* Bw   = w2 + (size_t)e * HH * DD;
    v8f acc[4] = {v8f{}, v8f{}, v8f{}, v8f{}};
    gemm_core_bf(Arow, Bw + n0, DD, HH, ldsb, acc);

    int n = lane & 15;
    int mbase = (lane < 16) ? 0 : 8;
    #pragma unroll
    for (int s = 0; s < 4; ++s) {
        int nc = n0 + s * 16 + n;
        float bb = b2[e * DD + nc];
        #pragma unroll
        for (int i = 0; i < 8; ++i) {
            int m = m0 + mbase + i;
            if (m < cnt) {
                int gr  = base + m;
                int tok = list[gr];
                float gt = glist[gr];
                atomicAdd(&hOut[(size_t)tok * DD + nc], gt * (acc[s][i] + bb));
            }
        }
    }
}

// ---------------------------------------------------------------- head GEMV
__global__ void head_gemv(const __bf16* __restrict__ r, const float* __restrict__ hw,
                          const float* __restrict__ hb, float* __restrict__ out)
{
    int idx = blockIdx.x * blockDim.x + threadIdx.x;
    if (idx >= BT * NCLSC) return;
    int c = idx % NCLSC;
    int t = idx / NCLSC;
    float s = hb[c];
    const __bf16* rr = r + (size_t)t * DD;
    for (int d = 0; d < DD; ++d) s += (float)rr[d] * hw[d * NCLSC + c];
    out[t * NCLSC + c] = s;
}

// ===========================================================================
extern "C" void kernel_launch(void* const* d_in, const int* in_sizes, int n_in,
                              void* d_out, int out_size, void* d_ws, size_t ws_size,
                              hipStream_t stream)
{
    const float* x        = (const float*)d_in[0];
    const float* conv1_w  = (const float*)d_in[1];
    const float* conv1_b  = (const float*)d_in[2];
    const float* bn1_g    = (const float*)d_in[3];
    const float* bn1_b    = (const float*)d_in[4];
    const float* bn1_m    = (const float*)d_in[5];
    const float* bn1_v    = (const float*)d_in[6];
    const float* conv2_w  = (const float*)d_in[7];
    const float* conv2_b  = (const float*)d_in[8];
    const float* bn2_g    = (const float*)d_in[9];
    const float* bn2_b    = (const float*)d_in[10];
    const float* bn2_m    = (const float*)d_in[11];
    const float* bn2_v    = (const float*)d_in[12];
    const float* fc_w     = (const float*)d_in[13];
    const float* fc_b     = (const float*)d_in[14];
    const float* ln_g     = (const float*)d_in[15];
    const float* ln_b     = (const float*)d_in[16];
    const float* router_w = (const float*)d_in[17];
    const float* router_b = (const float*)d_in[18];
    const float* w1       = (const float*)d_in[19];
    const float* b1       = (const float*)d_in[20];
    const float* w2       = (const float*)d_in[21];
    const float* b2       = (const float*)d_in[22];
    const float* hd_ln_g  = (const float*)d_in[23];
    const float* hd_ln_b  = (const float*)d_in[24];
    const float* head_w   = (const float*)d_in[25];
    const float* head_b   = (const float*)d_in[26];
    float* dout = (float*)d_out;

    // ---------------- workspace layout (bytes) ----------------
    char* wsb = (char*)d_ws;
    const size_t CONV1_BYTES = (size_t)BT * 64 * 32 * 32 * 4;   // 536,870,912
    float* f_conv1 = (float*)wsb;
    float* f_conv2 = (float*)(wsb + CONV1_BYTES);               // 1 GiB region
    // region A (conv1, dead after conv2) -> bf16 activations + small buffers
    size_t o = 0;
    __bf16* bf_pool = (__bf16*)(wsb + o); o += (size_t)BT * 2048 * 2;   //  8 MiB
    float*  f_hA    = (float*)(wsb + o);  o += (size_t)BT * DD * 4;     //  8 MiB
    float*  f_hB    = (float*)(wsb + o);  o += (size_t)BT * DD * 4;     //  8 MiB
    __bf16* bf_r    = (__bf16*)(wsb + o); o += (size_t)BT * DD * 2;     //  4 MiB
    float*  f_probs = (float*)(wsb + o);  o += (size_t)BT * EE * 4;
    __bf16* bf_ehg  = (__bf16*)(wsb + o); o += (size_t)2 * BT * HH * 2; // 16 MiB
    float*  f_gl    = (float*)(wsb + o);  o += (size_t)2 * BT * 4;
    float*  f_topg  = (float*)(wsb + o);  o += (size_t)2 * BT * 4;
    float*  f_prod  = (float*)(wsb + o);  o += 32;
    float*  f_aux   = (float*)(wsb + o);  o += 16;
    int*    i_topi  = (int*)(wsb + o);    o += (size_t)2 * BT * 4;
    int*    i_list  = (int*)(wsb + o);    o += (size_t)2 * BT * 4;
    int*    i_cnt   = (int*)(wsb + o);    o += 32;
    int*    i_off   = (int*)(wsb + o);    o += 32;
    int*    i_cur   = (int*)(wsb + o);    o += 32;
    // region B (conv2, dead after pooling) -> bf16 weights
    size_t ob = CONV1_BYTES;
    __bf16* bw_fc = (__bf16*)(wsb + ob); ob += (size_t)2048 * DD * 2;              // 4 MiB
    __bf16* bw1   = (__bf16*)(wsb + ob); ob += (size_t)NBLKC * EE * DD * HH * 2;   // 192 MiB
    __bf16* bw2   = (__bf16*)(wsb + ob); ob += (size_t)NBLKC * EE * HH * DD * 2;   // 192 MiB

    float* hcur = f_hA;
    float* hnxt = f_hB;

    init_aux_k<<<1, 1, 0, stream>>>(f_aux);

    // conv1 -> bn -> gelu
    {
        int total = BT * 64 * 32 * 32;
        conv_bn_gelu<<<(total + 255) / 256, 256, 0, stream>>>(
            x, conv1_w, conv1_b, bn1_g, bn1_b, bn1_m, bn1_v, f_conv1, BT, 3, 64);
    }
    // conv2 -> bn -> gelu
    {
        int total = BT * 128 * 32 * 32;
        conv_bn_gelu<<<(total + 255) / 256, 256, 0, stream>>>(
            f_conv1, conv2_w, conv2_b, bn2_g, bn2_b, bn2_m, bn2_v, f_conv2, BT, 64, 128);
    }
    // pool 8x8 -> bf16 flatten [BT, 2048]   (writes over dead conv1 region)
    {
        int total = BT * 128 * 4 * 4;
        avgpool_flatten<<<(total + 255) / 256, 256, 0, stream>>>(f_conv2, bf_pool, BT);
    }
    // one-time weight conversion to bf16 (into dead conv2 region)
    {
        long n4 = (long)2048 * DD / 4;
        cvt_bf16_k<<<(n4 + 255) / 256, 256, 0, stream>>>(fc_w, bw_fc, n4);
        n4 = (long)NBLKC * EE * DD * HH / 4;
        cvt_bf16_k<<<(n4 + 255) / 256, 256, 0, stream>>>(w1, bw1, n4);
        cvt_bf16_k<<<(n4 + 255) / 256, 256, 0, stream>>>(w2, bw2, n4);
    }
    // fc: h = gelu(pool @ fc_w + fc_b)   [2048,2048]x[2048,1024]  (WMMA)
    {
        dim3 grid(DD / 64, BT / 64);
        gemm_bias_gelu<<<grid, 128, 0, stream>>>(bf_pool, bw_fc, fc_b, hcur, BT, DD, 2048);
    }

    // ---------------- MoE blocks ----------------
    for (int i = 0; i < NBLKC; ++i) {
        // residual copy: hnxt = hcur (experts accumulate into it)
        hipMemcpyAsync(hnxt, hcur, (size_t)BT * DD * sizeof(float),
                       hipMemcpyDeviceToDevice, stream);
        layernorm_k<<<BT, 256, 0, stream>>>(hcur, ln_g + (size_t)i * DD,
                                            ln_b + (size_t)i * DD, bf_r, DD);
        router_k<<<BT / 256, 256, 0, stream>>>(bf_r, router_w + (size_t)i * DD * EE,
                                               router_b + (size_t)i * EE,
                                               f_probs, i_topi, f_topg);
        zero_counts_k<<<1, 32, 0, stream>>>(i_cnt);
        count_k<<<BT / 256, 256, 0, stream>>>(i_topi, i_cnt);
        scan_k<<<1, 1, 0, stream>>>(i_cnt, i_off, i_cur);
        scatter_k<<<BT / 256, 256, 0, stream>>>(i_topi, f_topg, i_cur, i_list, f_gl);
        aux_reduce_k<<<EE, 256, 0, stream>>>(f_probs, i_cnt, f_prod);
        aux_accum_k<<<1, 1, 0, stream>>>(f_prod, f_aux);

        const __bf16* w1i = bw1 + (size_t)i * EE * DD * HH;
        const float*  b1i = b1 + (size_t)i * EE * HH;
        const __bf16* w2i = bw2 + (size_t)i * EE * HH * DD;
        const float*  b2i = b2 + (size_t)i * EE * DD;

        dim3 g1(HH / 64, BT / 64, EE);   // worst-case tiles; uniform early-exit
        moe_gemm1<<<g1, 128, 0, stream>>>(bf_r, w1i, b1i, i_list, i_off, i_cnt, bf_ehg);
        dim3 g2(DD / 64, BT / 64, EE);
        moe_gemm2<<<g2, 128, 0, stream>>>(bf_ehg, w2i, b2i, i_list, f_gl, i_off, i_cnt, hnxt);

        float* t = hcur; hcur = hnxt; hnxt = t;
    }

    // ---------------- head ----------------
    layernorm_k<<<BT, 256, 0, stream>>>(hcur, hd_ln_g, hd_ln_b, bf_r, DD);
    head_gemv<<<(BT * NCLSC + 255) / 256, 256, 0, stream>>>(bf_r, head_w, head_b, dout);
    write_aux_k<<<1, 1, 0, stream>>>(f_aux, dout + BT * NCLSC);
}
